// infoNCE_68367289417783
// MI455X (gfx1250) — compile-verified, matrix operand order checked by reading, not run on hardware
//
#include <hip/hip_runtime.h>
#include <hip/hip_bf16.h>

// Problem constants (from reference setup_inputs)
#define DDIM  512      // feature dim d
#define BATCH 8192     // batch b
#define NNEG  8192     // negatives
#define TEMP_INV 0.01f
#define EPS   1e-12f

typedef __attribute__((ext_vector_type(16))) __bf16 v16bf;
typedef __attribute__((ext_vector_type(8)))  float  v8f;

union Frag {
    uint4 q[2];
    v16bf v;
};

__device__ __forceinline__ unsigned short f2bf(float f) {
    union { float f; unsigned int u; } c;
    c.f = f;
    unsigned int u = c.u;
    // round-to-nearest-even bf16
    u += 0x7FFFu + ((u >> 16) & 1u);
    return (unsigned short)(u >> 16);
}

// ===========================================================================
// FAST PATH (needs ~151 MB workspace for bf16 transposed copies)
// ===========================================================================

// ---------------------------------------------------------------------------
// Transpose + convert: src (BATCH, N) f32 row-major -> dst (N, BATCH) bf16.
// 64x64 tiles through LDS; coalesced f32 reads, 32B-vector bf16 writes.
// grid = (BATCH/64, N/64), 256 threads.
// ---------------------------------------------------------------------------
__global__ __launch_bounds__(256)
void convert_transpose_kernel(const float* __restrict__ src,
                              unsigned short* __restrict__ dst,
                              int N) {
    __shared__ __align__(16) unsigned short tile[64][72];  // [n][b], 144B rows
    const int b0 = blockIdx.x * 64;
    const int n0 = blockIdx.y * 64;
    const int t  = threadIdx.x;

    const int nc = t & 63;   // n within tile
    const int br = t >> 6;   // 0..3 base b row
    #pragma unroll
    for (int i = 0; i < 16; ++i) {
        const int bs = br + i * 4;
        tile[nc][bs] = f2bf(src[(size_t)(b0 + bs) * N + (n0 + nc)]);
    }
    __syncthreads();

    const int nr = t >> 2;          // 0..63 output row (n)
    const int qs = (t & 3) * 16;    // 0/16/32/48 halves
    const uint4* s = (const uint4*)&tile[nr][qs];
    uint4* d = (uint4*)&dst[(size_t)(n0 + nr) * BATCH + b0 + qs];
    d[0] = s[0];
    d[1] = s[1];
}

// ---------------------------------------------------------------------------
// bf16 GEMM column-norm^2 kernel. At = ipt^T (DDIM, BATCH) bf16 row-major.
// Bt = X^T (NX, BATCH) bf16 row-major. out2[j] += ||column j||^2 partials.
// Workgroup: 64 k-rows x 256 j-cols. Wave w owns j-tiles {2w,2w+1} x 4
// k-tiles -> 8 WMMAs/chunk with each A fragment reused twice.
// K (=BATCH) in chunks of 32, staged to LDS with
// global_load_async_to_lds_b128 (ASYNCcnt), double-buffered.
// grid = (DDIM/64, NX/256), 256 threads (8 waves).
// ---------------------------------------------------------------------------
__global__ __launch_bounds__(256)
void norm2_gemm_bf16_kernel(const unsigned short* __restrict__ At,
                            const unsigned short* __restrict__ Bt,
                            float* __restrict__ out2) {
    // rows 0..63 = A k-rows, rows 64..319 = B j-rows; stride 40 halves (80 B)
    __shared__ __align__(16) unsigned short lds[2][320 * 40];

    const int k0   = blockIdx.x * 64;
    const int j0   = blockIdx.y * 256;
    const int t    = threadIdx.x;
    const int wave = t >> 5;
    const int lane = t & 31;
    const int lr   = lane & 15;
    const int hi   = lane >> 4;

    // 5 async staging tasks per thread: 320 rows x 4 16-byte segments = 1280
    unsigned long long gbase[5];
    unsigned ldsrel[5];
    #pragma unroll
    for (int s = 0; s < 5; ++s) {
        const int tid = t + 256 * s;
        const int seg = tid & 3;        // 16-byte segment within 64-B row
        const int row = tid >> 2;       // 0..319
        const unsigned short* g = (row < 64)
            ? (At + (size_t)(k0 + row) * BATCH)
            : (Bt + (size_t)(j0 + (row - 64)) * BATCH);
        gbase[s]  = (unsigned long long)(uintptr_t)g + (unsigned)(seg * 16);
        ldsrel[s] = (unsigned)(row * 80 + seg * 16);
    }
    const unsigned lbase0 = (unsigned)(uintptr_t)(void*)&lds[0][0];
    const unsigned lbase1 = (unsigned)(uintptr_t)(void*)&lds[1][0];

    v8f acc[4][2];
    #pragma unroll
    for (int kt = 0; kt < 4; ++kt) {
        acc[kt][0] = v8f{};
        acc[kt][1] = v8f{};
    }

    // prologue: chunk 0 -> buffer 0
    #pragma unroll
    for (int s = 0; s < 5; ++s) {
        const unsigned long long ga = gbase[s];
        const unsigned la = lbase0 + ldsrel[s];
        asm volatile("global_load_async_to_lds_b128 %0, %1, off"
                     :: "v"(la), "v"(ga) : "memory");
    }

    const int NCH = BATCH / 32;  // 256 chunks
    for (int c = 0; c < NCH; ++c) {
        if (c + 1 < NCH) {
            // issue next chunk into the other buffer, then wait for current
            const unsigned long long boff =
                (unsigned long long)(c + 1) * 32ull * 2ull;
            const unsigned nb = ((c + 1) & 1) ? lbase1 : lbase0;
            #pragma unroll
            for (int s = 0; s < 5; ++s) {
                const unsigned long long ga = gbase[s] + boff;
                const unsigned la = nb + ldsrel[s];
                asm volatile("global_load_async_to_lds_b128 %0, %1, off"
                             :: "v"(la), "v"(ga) : "memory");
            }
            asm volatile("s_wait_asynccnt 0x5" ::: "memory");
        } else {
            asm volatile("s_wait_asynccnt 0x0" ::: "memory");
        }
        __syncthreads();

        const unsigned short* base = (c & 1) ? &lds[1][0] : &lds[0][0];

        // Two B fragments (32x16): j-tiles 2*wave and 2*wave+1
        Frag fb[2];
        #pragma unroll
        for (int jt = 0; jt < 2; ++jt) {
            const unsigned short* rowB =
                base + (64 + wave * 32 + jt * 16 + lr) * 40;
            fb[jt].q[0] = *(const uint4*)(rowB + hi * 8);
            fb[jt].q[1] = *(const uint4*)(rowB + 16 + hi * 8);
        }
        // 4 A fragments, each reused for both j-tiles -> 8 WMMAs
        #pragma unroll
        for (int kt = 0; kt < 4; ++kt) {
            Frag fa;
            const unsigned short* rowA = base + (kt * 16 + lr) * 40;
            fa.q[0] = *(const uint4*)(rowA + hi * 8);
            fa.q[1] = *(const uint4*)(rowA + 16 + hi * 8);
            acc[kt][0] = __builtin_amdgcn_wmma_f32_16x16x32_bf16(
                false, fa.v, false, fb[0].v, (short)0, acc[kt][0], false, false);
            acc[kt][1] = __builtin_amdgcn_wmma_f32_16x16x32_bf16(
                false, fa.v, false, fb[1].v, (short)0, acc[kt][1], false, false);
        }
        __syncthreads();
    }

    #pragma unroll
    for (int jt = 0; jt < 2; ++jt) {
        float s = 0.0f;
        #pragma unroll
        for (int kt = 0; kt < 4; ++kt) {
            #pragma unroll
            for (int r = 0; r < 8; ++r) {
                s += acc[kt][jt][r] * acc[kt][jt][r];
            }
        }
        s += __shfl_xor(s, 16);
        if (lane < 16) {
            atomicAdd(&out2[j0 + wave * 32 + jt * 16 + lane], s);
        }
    }
}

// ===========================================================================
// FALLBACK PATH (fused f32->bf16 staging; no big workspace needed)
// ===========================================================================
__global__ __launch_bounds__(256)
void norm2_gemm_fused_kernel(const float* __restrict__ ipt,
                             const float* __restrict__ X,
                             int NX,
                             float* __restrict__ out2) {
    __shared__ __align__(16) unsigned short lA[64 * 40];   // [k][b]
    __shared__ __align__(16) unsigned short lB[128 * 40];  // [j][b]

    const int k0   = blockIdx.x * 64;
    const int j0   = blockIdx.y * 128;
    const int t    = threadIdx.x;
    const int wave = t >> 5;
    const int lane = t & 31;
    const int lr   = lane & 15;
    const int hi   = lane >> 4;

    const int kk    = t & 63;
    const int brow  = t >> 6;
    const int jj    = t & 127;
    const int brow2 = t >> 7;

    v8f acc[4];
    acc[0] = v8f{}; acc[1] = v8f{}; acc[2] = v8f{}; acc[3] = v8f{};

    for (int b0 = 0; b0 < BATCH; b0 += 32) {
        #pragma unroll
        for (int i = 0; i < 8; ++i) {
            const int bs = brow + i * 4;
            lA[kk * 40 + bs] = f2bf(ipt[(size_t)(b0 + bs) * DDIM + (k0 + kk)]);
        }
        #pragma unroll
        for (int i = 0; i < 16; ++i) {
            const int bs = brow2 + i * 2;
            lB[jj * 40 + bs] = f2bf(X[(size_t)(b0 + bs) * NX + (j0 + jj)]);
        }
        __syncthreads();

        Frag fb;
        {
            const unsigned short* rowB = &lB[(wave * 16 + lr) * 40];
            fb.q[0] = *(const uint4*)(rowB + hi * 8);
            fb.q[1] = *(const uint4*)(rowB + 16 + hi * 8);
        }
        #pragma unroll
        for (int kt = 0; kt < 4; ++kt) {
            Frag fa;
            const unsigned short* rowA = &lA[(kt * 16 + lr) * 40];
            fa.q[0] = *(const uint4*)(rowA + hi * 8);
            fa.q[1] = *(const uint4*)(rowA + 16 + hi * 8);
            acc[kt] = __builtin_amdgcn_wmma_f32_16x16x32_bf16(
                false, fa.v, false, fb.v, (short)0, acc[kt], false, false);
        }
        __syncthreads();
    }

    float s = 0.0f;
    #pragma unroll
    for (int r = 0; r < 8; ++r) {
        s += acc[0][r] * acc[0][r];
        s += acc[1][r] * acc[1][r];
        s += acc[2][r] * acc[2][r];
        s += acc[3][r] * acc[3][r];
    }
    s += __shfl_xor(s, 16);
    if (lane < 16) {
        atomicAdd(&out2[j0 + wave * 16 + lane], s);
    }
}

// ===========================================================================
// Shared small kernels
// ===========================================================================
__global__ void init_zero_kernel(float* p, int n) {
    int i = blockIdx.x * blockDim.x + threadIdx.x;
    if (i < n) p[i] = 0.0f;
}

__global__ __launch_bounds__(256)
void finalize_kernel(const float* __restrict__ pos2,
                     const float* __restrict__ neg2,
                     float* __restrict__ out) {
    __shared__ float shA[256];
    __shared__ float shB[256];
    const int t = threadIdx.x;

    float sumA = 0.0f, sumL = 0.0f;
    for (int k = t; k < NNEG; k += 256) {
        const float a = __expf(sqrtf(neg2[k]) * TEMP_INV);
        sumA += a;
        sumL += __logf(a + EPS);
    }
    shA[t] = sumA;
    shB[t] = sumL;
    __syncthreads();
    for (int off = 128; off > 0; off >>= 1) {
        if (t < off) { shA[t] += shA[t + off]; shB[t] += shB[t + off]; }
        __syncthreads();
    }
    const float S = shA[0];
    const float L = shB[0];
    __syncthreads();

    float sumLn = 0.0f;
    for (int i = t; i < DDIM; i += 256) {
        const float ap = __expf(sqrtf(pos2[i]) * TEMP_INV);
        const float denom = ap + S + EPS;
        sumLn += __logf(ap + EPS) + L - (float)(NNEG + 1) * __logf(denom);
    }
    shA[t] = sumLn;
    __syncthreads();
    for (int off = 128; off > 0; off >>= 1) {
        if (t < off) shA[t] += shA[t + off];
        __syncthreads();
    }
    if (t == 0) {
        const float Ln = shA[0] / (float)(NNEG + 1);
        out[0] = -Ln / (float)DDIM;
    }
}

// ===========================================================================
extern "C" void kernel_launch(void* const* d_in, const int* in_sizes, int n_in,
                              void* d_out, int out_size, void* d_ws, size_t ws_size,
                              hipStream_t stream) {
    (void)in_sizes; (void)n_in; (void)out_size;

    const float* opt = (const float*)d_in[0];   // (8192, 512)
    const float* ipt = (const float*)d_in[1];   // (8192, 512)
    const float* neg = (const float*)d_in[2];   // (8192, 8192)

    float* pos2 = (float*)d_ws;                 // 512 f32
    float* neg2 = pos2 + DDIM;                  // 8192 f32

    const int ninit = DDIM + NNEG;
    init_zero_kernel<<<(ninit + 255) / 256, 256, 0, stream>>>(pos2, ninit);

    // bf16 transposed copies live after a 64 KB header
    const size_t hdr  = 65536;
    const size_t rows = (size_t)DDIM + (size_t)DDIM + (size_t)NNEG; // iptT+optT+negT
    const size_t need = hdr + rows * (size_t)BATCH * 2;

    if (ws_size >= need) {
        unsigned short* iptT = (unsigned short*)((char*)d_ws + hdr);
        unsigned short* optT = iptT + (size_t)DDIM * BATCH;
        unsigned short* negT = optT + (size_t)DDIM * BATCH;

        convert_transpose_kernel<<<dim3(BATCH / 64, DDIM / 64), 256, 0, stream>>>(
            ipt, iptT, DDIM);
        convert_transpose_kernel<<<dim3(BATCH / 64, DDIM / 64), 256, 0, stream>>>(
            opt, optT, DDIM);
        convert_transpose_kernel<<<dim3(BATCH / 64, NNEG / 64), 256, 0, stream>>>(
            neg, negT, NNEG);

        norm2_gemm_bf16_kernel<<<dim3(DDIM / 64, DDIM / 256), 256, 0, stream>>>(
            iptT, optT, pos2);
        norm2_gemm_bf16_kernel<<<dim3(DDIM / 64, NNEG / 256), 256, 0, stream>>>(
            iptT, negT, neg2);
    } else {
        norm2_gemm_fused_kernel<<<dim3(DDIM / 64, DDIM / 128), 256, 0, stream>>>(
            ipt, opt, DDIM, pos2);
        norm2_gemm_fused_kernel<<<dim3(DDIM / 64, NNEG / 128), 256, 0, stream>>>(
            ipt, neg, NNEG, neg2);
    }

    finalize_kernel<<<1, 256, 0, stream>>>(pos2, neg2, (float*)d_out);
}
